// RITSModel_17987323035685
// MI455X (gfx1250) — compile-verified, matrix operand order checked by reading, not run on hardware
//
#include <hip/hip_runtime.h>

// ---------------- Types ----------------
typedef __attribute__((ext_vector_type(16))) _Float16 v16h;
typedef __attribute__((ext_vector_type(8)))  float    v8f;
typedef __attribute__((ext_vector_type(4)))  unsigned v4u;

#define GLOBAL_AS __attribute__((address_space(1)))
typedef const _Float16 GLOBAL_AS* gwp;   // global-addrspace weight pointer

// ---------------- Problem constants ----------------
#define BB 512
#define TT 128
#define DD 128
#define HHN 256
#define FRAG_HW 512  // halfwords per packed B fragment (32 lanes * 16 f16)

// Packed-weight offsets in halfwords inside d_ws.
// frag counts: DH=(256/16)*(128/32)=64, HR=64, FR=32, WC=64, IH=512, HH=512
#define OFF_DH 0
#define OFF_HR (64 * FRAG_HW)
#define OFF_FR (128 * FRAG_HW)
#define OFF_WC (160 * FRAG_HW)
#define OFF_IH (224 * FRAG_HW)
#define OFF_HH (736 * FRAG_HW)

// ---------------- Helpers ----------------
__device__ __forceinline__ v8f zero8() {
  v8f z = {0.f, 0.f, 0.f, 0.f, 0.f, 0.f, 0.f, 0.f};
  return z;
}
__device__ __forceinline__ float sigm_(float x) { return 1.f / (1.f + __expf(-x)); }
__device__ __forceinline__ float tanh_(float x) {
  float e = __expf(2.f * x);
  return (e - 1.f) / (e + 1.f);
}

// Opaque pointer launder in the GLOBAL address space: defeats LICM (so weight
// fragments stream from L2 inside the t-loop instead of being hoisted+spilled)
// while keeping addrspace(1) so the loads lower to global_load_b128 (LOADcnt
// only — decoupled from the LDS DScnt waits), not flat_load.
__device__ __forceinline__ gwp launder_ptr(const _Float16* p) {
  gwp g = (gwp)p;
  asm volatile("" : "+s"(g));
  return g;
}

// A-fragment (16x32 f16, MxK) gathered from an LDS row-major panel.
// Lane l: m = l&15, half = l>>4; pair p holds K = ks*32 + kbase(p,half) + {0,1}
__device__ __forceinline__ v16h load_a_lds(const _Float16* base, int stride, int ks, int lane) {
  const int mrow = lane & 15;
  const int hf   = lane >> 4;
  union { v16h v; unsigned u[8]; } r;
  const _Float16* row = base + mrow * stride + ks * 32;
#pragma unroll
  for (int p = 0; p < 8; ++p) {
    int k = ((p < 4) ? 0 : 16) + hf * 8 + ((p & 3) << 1);
    r.u[p] = *(const unsigned*)(row + k);   // two consecutive f16 -> one ds_load_b32
  }
  return r.v;
}

// B-fragment (32x16 f16, KxN) pre-packed contiguously: frag base + lane*16 halfwords.
__device__ __forceinline__ v16h load_b_glb(gwp w, int frag, int lane) {
  gwp p = w + (size_t)frag * FRAG_HW + lane * 16;
  union { v16h v; v4u q[2]; } r;
  r.q[0] = *(const v4u GLOBAL_AS*)p;        // global_load_b128
  r.q[1] = *(const v4u GLOBAL_AS*)(p + 8);  // global_load_b128
  return r.v;
}

// ---------------- Weight pre-pack: W[N,K] f32 -> WMMA B-fragment f16 layout ----------------
__global__ void pack_weights_kernel(const float* __restrict__ W, _Float16* __restrict__ dst,
                                    int N, int K, int maskDiag) {
  int tid = blockIdx.x * blockDim.x + threadIdx.x;
  int KS = K >> 5;
  int total = (N >> 4) * KS * 32;
  if (tid >= total) return;
  int lane = tid & 31, frag = tid >> 5;
  int nt = frag / KS, ks = frag % KS;
  int n = nt * 16 + (lane & 15);
  int hf = lane >> 4;
  _Float16* o = dst + (size_t)frag * FRAG_HW + lane * 16;
#pragma unroll
  for (int p = 0; p < 8; ++p) {
    int kb = ks * 32 + ((p < 4) ? 0 : 16) + hf * 8 + ((p & 3) << 1);
#pragma unroll
    for (int j = 0; j < 2; ++j) {
      int k = kb + j;
      float v = W[(size_t)n * K + k];
      if (maskDiag && n == k) v = 0.f;   // zero-diagonal for FeatureRegression
      o[p * 2 + j] = (_Float16)v;
    }
  }
}

// ---------------- Persistent recurrent kernel: one WG per 16 batch rows ----------------
__global__ __launch_bounds__(256, 1) void rits_recurrent_kernel(
    const float* __restrict__ x, const int* __restrict__ mask,
    const float* __restrict__ b_ih, const float* __restrict__ b_hh,
    const float* __restrict__ b_dh, const float* __restrict__ W_dx,
    const float* __restrict__ b_dx, const float* __restrict__ b_hr,
    const float* __restrict__ b_fr, const float* __restrict__ b_wc,
    const _Float16* __restrict__ wpk, float* __restrict__ out) {
  __shared__ float    cS[16][HHN];       // LSTM cell state, fp32 (16 KB)
  __shared__ _Float16 hS[16][HHN];       // hidden state, f16 GEMM operand (8 KB)
  __shared__ _Float16 in2[16][2 * DD];   // concat panel: [gamma_x|m] then [c_c|m] (8 KB)
  __shared__ _Float16 dS[16][DD];        // delta (exact small ints in f16) (4 KB)
  __shared__ _Float16 mS[16][DD];        // current mask (4 KB)
  __shared__ _Float16 xS[16][DD];        // x_t (4 KB)
  __shared__ _Float16 xhS[16][DD];       // x_h (4 KB)
  __shared__ _Float16 zhS[16][DD];       // z_h (4 KB)
  __shared__ _Float16 xcS[16][DD];       // x_c (4 KB)

  const int tid = threadIdx.x;
  const int lane = tid & 31;
  const int wv = tid >> 5;          // 8 waves
  const int hf = lane >> 4;
  const int nn = lane & 15;
  const int bbase = blockIdx.x * 16;

  for (int i = tid; i < 16 * HHN; i += 256) {
    cS[i >> 8][i & 255] = 0.f;
    hS[i >> 8][i & 255] = (_Float16)0.f;
  }
  for (int i = tid; i < 16 * DD; i += 256) {
    dS[i >> 7][i & 127] = (_Float16)0.f;
    mS[i >> 7][i & 127] = (_Float16)0.f;
  }
  __syncthreads();

  const size_t BTD = (size_t)BB * TT * DD;
  float* out1 = out;
  float* out2 = out + BTD;
  float* out3 = out + 2 * BTD;
  float* out4 = out + 3 * BTD;

  for (int t = 0; t < TT; ++t) {
    // Re-laundered every step: all weight-fragment addresses become loop-variant,
    // so the compiler streams them from L2 instead of hoisting + spilling.
    gwp w_t = launder_ptr(wpk);

    // S1: delta[t] = m[t-1] + (1-m[t-1]) * (delta[t-1] + 1)   (delta[0] = 0)
    if (t > 0) {
      for (int i = tid; i < 16 * DD; i += 256) {
        int r = i >> 7, c = i & 127;
        float mp = (float)mS[r][c], dp = (float)dS[r][c];
        dS[r][c] = (_Float16)(mp + (1.f - mp) * (dp + 1.f));
      }
      __syncthreads();
    }
    // S2: stage x_t and m_t (vectorized: 512 float4/int4 tiles, 2 per thread)
    for (int i = tid; i < 16 * (DD / 4); i += 256) {
      int r = i >> 5, c4 = (i & 31) * 4;
      size_t g = ((size_t)(bbase + r) * TT + t) * DD + c4;
      float4 xv = *(const float4*)(x + g);
      int4   mv = *(const int4*)(mask + g);
      xS[r][c4 + 0] = (_Float16)xv.x;  xS[r][c4 + 1] = (_Float16)xv.y;
      xS[r][c4 + 2] = (_Float16)xv.z;  xS[r][c4 + 3] = (_Float16)xv.w;
      mS[r][c4 + 0] = (_Float16)(float)mv.x;  mS[r][c4 + 1] = (_Float16)(float)mv.y;
      mS[r][c4 + 2] = (_Float16)(float)mv.z;  mS[r][c4 + 3] = (_Float16)(float)mv.w;
    }
    __syncthreads();

    // S3: gamma_h = exp(-relu(dt @ W_dh^T + b_dh)); h *= gamma_h.  16 N-tiles, 2 per wave.
#pragma unroll
    for (int j = 0; j < 2; ++j) {
      int nt = wv * 2 + j;
      v8f acc = zero8();
#pragma unroll
      for (int ks = 0; ks < 4; ++ks) {
        v16h a = load_a_lds(&dS[0][0], DD, ks, lane);
        v16h b = load_b_glb(w_t + OFF_DH, nt * 4 + ks, lane);
        acc = __builtin_amdgcn_wmma_f32_16x16x32_f16(false, a, false, b, (short)0, acc, false, false);
      }
      int u = nt * 16 + nn;
      float bias = b_dh[u];
#pragma unroll
      for (int r = 0; r < 8; ++r) {
        int mr = r + hf * 8;
        float g = acc[r] + bias;
        g = g > 0.f ? g : 0.f;
        hS[mr][u] = (_Float16)((float)hS[mr][u] * __expf(-g));
      }
    }
    __syncthreads();

    // S4: x_h = h @ W_hr^T + b_hr (tile wv); plus gamma_x elementwise into in2=[gamma_x|m]
    {
      v8f acc = zero8();
#pragma unroll
      for (int ks = 0; ks < 8; ++ks) {
        v16h a = load_a_lds(&hS[0][0], HHN, ks, lane);
        v16h b = load_b_glb(w_t + OFF_HR, wv * 8 + ks, lane);
        acc = __builtin_amdgcn_wmma_f32_16x16x32_f16(false, a, false, b, (short)0, acc, false, false);
      }
      int u = wv * 16 + nn;
      float bias = b_hr[u];
#pragma unroll
      for (int r = 0; r < 8; ++r) {
        int mr = r + hf * 8;
        float xh = acc[r] + bias;
        out1[((size_t)(bbase + mr) * TT + t) * DD + u] = xh;
        xhS[mr][u] = (_Float16)xh;
      }
      for (int i = tid; i < 16 * DD; i += 256) {
        int r = i >> 7, c = i & 127;
        float dv = (float)dS[r][c];
        float g = dv * W_dx[c * DD + c] + b_dx[c];
        g = g > 0.f ? g : 0.f;
        in2[r][c] = (_Float16)__expf(-g);
        in2[r][DD + c] = mS[r][c];
      }
    }
    __syncthreads();

    // S5: x_c = m ? x : x_h
    for (int i = tid; i < 16 * DD; i += 256) {
      int r = i >> 7, c = i & 127;
      xcS[r][c] = ((float)mS[r][c] > 0.f) ? xS[r][c] : xhS[r][c];
    }
    __syncthreads();

    // S6: z_h = x_c @ W_fr_masked^T + b_fr (tile wv)
    {
      v8f acc = zero8();
#pragma unroll
      for (int ks = 0; ks < 4; ++ks) {
        v16h a = load_a_lds(&xcS[0][0], DD, ks, lane);
        v16h b = load_b_glb(w_t + OFF_FR, wv * 4 + ks, lane);
        acc = __builtin_amdgcn_wmma_f32_16x16x32_f16(false, a, false, b, (short)0, acc, false, false);
      }
      int u = wv * 16 + nn;
      float bias = b_fr[u];
#pragma unroll
      for (int r = 0; r < 8; ++r) {
        int mr = r + hf * 8;
        float zh = acc[r] + bias;
        out2[((size_t)(bbase + mr) * TT + t) * DD + u] = zh;
        zhS[mr][u] = (_Float16)zh;
      }
    }
    __syncthreads();

    // S7a: alpha = [gamma_x|m] @ W_wc^T + b_wc; c_h = alpha*z_h + (1-alpha)*x_h
    float chv[8];
    {
      v8f acc = zero8();
#pragma unroll
      for (int ks = 0; ks < 8; ++ks) {
        v16h a = load_a_lds(&in2[0][0], 2 * DD, ks, lane);
        v16h b = load_b_glb(w_t + OFF_WC, wv * 8 + ks, lane);
        acc = __builtin_amdgcn_wmma_f32_16x16x32_f16(false, a, false, b, (short)0, acc, false, false);
      }
      int u = wv * 16 + nn;
      float bias = b_wc[u];
#pragma unroll
      for (int r = 0; r < 8; ++r) {
        int mr = r + hf * 8;
        float al = acc[r] + bias;
        float ch = al * (float)zhS[mr][u] + (1.f - al) * (float)xhS[mr][u];
        chv[r] = ch;
        size_t o = ((size_t)(bbase + mr) * TT + t) * DD + u;
        out3[o] = ch;
        out4[o] = ch;
      }
    }
    __syncthreads();

    // S7b: lstm_in = [c_c|m]; overwrite low half of in2 with c_c = m ? x : c_h
    {
      int u = wv * 16 + nn;
#pragma unroll
      for (int r = 0; r < 8; ++r) {
        int mr = r + hf * 8;
        float cc = ((float)mS[mr][u] > 0.f) ? (float)xS[mr][u] : chv[r];
        in2[mr][u] = (_Float16)cc;
      }
    }
    __syncthreads();

    // S8: gates = lstm_in @ W_ih^T + h @ W_hh^T + b.  64 N-tiles; wave w owns
    // tiles n = gi*16 + (w, w+8) so i/f/g/o of the same units stay in-register.
    {
      v8f acc[4][2];
#pragma unroll
      for (int gi = 0; gi < 4; ++gi)
#pragma unroll
        for (int q = 0; q < 2; ++q) acc[gi][q] = zero8();

#pragma unroll
      for (int ks = 0; ks < 8; ++ks) {
        v16h a = load_a_lds(&in2[0][0], 2 * DD, ks, lane);
#pragma unroll
        for (int gi = 0; gi < 4; ++gi)
#pragma unroll
          for (int q = 0; q < 2; ++q) {
            int n = gi * 16 + wv + q * 8;
            v16h b = load_b_glb(w_t + OFF_IH, n * 8 + ks, lane);
            acc[gi][q] = __builtin_amdgcn_wmma_f32_16x16x32_f16(false, a, false, b, (short)0,
                                                                acc[gi][q], false, false);
          }
      }
#pragma unroll
      for (int ks = 0; ks < 8; ++ks) {
        v16h a = load_a_lds(&hS[0][0], HHN, ks, lane);
#pragma unroll
        for (int gi = 0; gi < 4; ++gi)
#pragma unroll
          for (int q = 0; q < 2; ++q) {
            int n = gi * 16 + wv + q * 8;
            v16h b = load_b_glb(w_t + OFF_HH, n * 8 + ks, lane);
            acc[gi][q] = __builtin_amdgcn_wmma_f32_16x16x32_f16(false, a, false, b, (short)0,
                                                                acc[gi][q], false, false);
          }
      }
      __syncthreads();  // everyone done reading hS/in2 before state write-back

#pragma unroll
      for (int q = 0; q < 2; ++q) {
        int u = (wv + q * 8) * 16 + nn;
        float bi[4];
#pragma unroll
        for (int gi = 0; gi < 4; ++gi) bi[gi] = b_ih[gi * 256 + u] + b_hh[gi * 256 + u];
#pragma unroll
        for (int r = 0; r < 8; ++r) {
          int mr = r + hf * 8;
          float iv = acc[0][q][r] + bi[0];
          float fv = acc[1][q][r] + bi[1];
          float gv = acc[2][q][r] + bi[2];
          float ov = acc[3][q][r] + bi[3];
          float co = cS[mr][u];
          float cn = sigm_(fv) * co + sigm_(iv) * tanh_(gv);
          cS[mr][u] = cn;
          hS[mr][u] = (_Float16)(sigm_(ov) * tanh_(cn));
        }
      }
    }
    __syncthreads();
  }
}

// ---------------- Host launch ----------------
extern "C" void kernel_launch(void* const* d_in, const int* in_sizes, int n_in,
                              void* d_out, int out_size, void* d_ws, size_t ws_size,
                              hipStream_t stream) {
  (void)in_sizes; (void)n_in; (void)out_size; (void)ws_size;
  const float* x    = (const float*)d_in[0];
  const int*   mk   = (const int*)d_in[1];
  const float* W_ih = (const float*)d_in[2];
  const float* W_hh = (const float*)d_in[3];
  const float* b_ih = (const float*)d_in[4];
  const float* b_hh = (const float*)d_in[5];
  const float* W_dh = (const float*)d_in[6];
  const float* b_dh = (const float*)d_in[7];
  const float* W_dx = (const float*)d_in[8];
  const float* b_dx = (const float*)d_in[9];
  const float* W_hr = (const float*)d_in[10];
  const float* b_hr = (const float*)d_in[11];
  const float* W_fr = (const float*)d_in[12];
  const float* b_fr = (const float*)d_in[13];
  const float* W_wc = (const float*)d_in[14];
  const float* b_wc = (const float*)d_in[15];

  _Float16* wpk = (_Float16*)d_ws;   // ~1.25 MB of packed f16 weights

  auto pack = [&](const float* W, int off, int N, int K, int md) {
    int total = (N / 16) * (K / 32) * 32;
    pack_weights_kernel<<<(total + 255) / 256, 256, 0, stream>>>(W, wpk + off, N, K, md);
  };
  pack(W_dh, OFF_DH, 256, 128, 0);
  pack(W_hr, OFF_HR, 128, 256, 0);
  pack(W_fr, OFF_FR, 128, 128, 1);   // zero diagonal
  pack(W_wc, OFF_WC, 128, 256, 0);
  pack(W_ih, OFF_IH, 1024, 256, 0);
  pack(W_hh, OFF_HH, 1024, 256, 0);

  rits_recurrent_kernel<<<BB / 16, 256, 0, stream>>>(
      x, mk, b_ih, b_hh, b_dh, W_dx, b_dx, b_hr, b_fr, b_wc, wpk, (float*)d_out);
}
